// EfficientOptTransformer_16604343566488
// MI455X (gfx1250) — matrix operand order repaired and checked by reading, not compiled, not run on hardware
//
#include <hip/hip_runtime.h>
#include <hip/hip_bf16.h>
#include <math.h>

// Problem constants (match reference)
#define BB   1024
#define NN   100
#define DD   128
#define HH   8
#define DKK  16
#define LL   3
#define HFF_ 512
#define BN_  (BB * NN)
#define NEGV (-1000000000.0f)
#define EPS_ 1e-5f

typedef __attribute__((ext_vector_type(16))) __bf16          v16bf;
typedef __attribute__((ext_vector_type(16))) unsigned short  v16us;
typedef __attribute__((ext_vector_type(8)))  float           v8f;
typedef __attribute__((ext_vector_type(4)))  unsigned short  us4;

__device__ __forceinline__ unsigned short f2bf(float f) {
  union { float f; unsigned u; } x; x.f = f;
  unsigned r = x.u + 0x7FFFu + ((x.u >> 16) & 1u);   // round-to-nearest-even
  return (unsigned short)(r >> 16);
}

// ---------------------------------------------------------------------------
// One-time weight convert+transpose: dst[c*K + k] = bf16(src[k*Nc + c]).
// With W^T stored [Nc,K], a WMMA B-fragment per lane is two contiguous 8-u16 runs.
// ---------------------------------------------------------------------------
__global__ void convT_kernel(const float* __restrict__ src,
                             unsigned short* __restrict__ dst, int K, int Nc)
{
  int idx = blockIdx.x * 256 + threadIdx.x;
  if (idx >= K * Nc) return;
  int k = idx / Nc, c = idx - k * Nc;
  dst[(size_t)c * K + k] = f2bf(src[idx]);
}

// ---------------------------------------------------------------------------
// WMMA GEMM: C[M,Nc] = act(A[M,K] @ W[K,Nc] + bias) (+ Res), f32 accumulate.
// A f32 row-major; W pre-converted bf16 TRANSPOSED [Nc,K]; C/Res f32.
// Block = 256 threads = 8 waves; tile = 16 rows x 128 cols, one 16x16 WMMA
// tile per wave, K chunked by 32 (v_wmma_f32_16x16x32_bf16), fully unrolled.
// Fragment layouts (ISA 7.12.2, 16-bit):
//   A 16x32: lane<16 -> k half 0, lane>=16 -> +8; per lane two contiguous
//            8-element runs [hA, hA+7] and [hA+16, hA+23]  -> 2x ds b128
//   B 32x16: same k order as A: runs [hB, hB+7] and [hB+16, hB+23] of one
//            column -> 2x global b128 from W^T row
//   C f32:   VGPR r -> row r (lanes 0-15) / 8+r (lanes 16-31), col = lane&15
// ---------------------------------------------------------------------------
template <int K, bool BIAS, bool RELU, bool RES>
__global__ __launch_bounds__(256)
void gemm_wmma(const float* __restrict__ A, const unsigned short* __restrict__ Wt,
               const float* __restrict__ bias, const float* __restrict__ Res,
               float* __restrict__ C, int Ncols)
{
  __shared__ unsigned short a_s[16 * K];
  const int tid  = threadIdx.x;
  const int wave = tid >> 5, lane = tid & 31;
  const int m0 = blockIdx.x << 4;
  const int n0 = blockIdx.y << 7;                       // 8 waves * 16 cols

  // Stage A strip (16 contiguous rows => contiguous memory): f32x4 -> bf16x4
  {
    const float* a0 = A + (size_t)m0 * K;
#pragma unroll
    for (int i = tid * 4; i < 16 * K; i += 1024) {
      const float4 f = *(const float4*)(a0 + i);
      us4 h = { f2bf(f.x), f2bf(f.y), f2bf(f.z), f2bf(f.w) };
      *(us4*)(a_s + i) = h;
    }
  }
  __syncthreads();

  const int row = lane & 15;
  const int col = n0 + (wave << 4) + row;
  const int hA  = (lane & 16) ? 8  : 0;
  const int hB  = (lane & 16) ? 8  : 0;
  const unsigned short* wrow = Wt + (size_t)col * K;

  v8f acc = {0.f, 0.f, 0.f, 0.f, 0.f, 0.f, 0.f, 0.f};
#pragma unroll
  for (int kk = 0; kk < K; kk += 32) {
    union { uint4 q[2]; v16us v; } af, bf;
    af.q[0] = *(const uint4*)(a_s + row * K + kk + hA);
    af.q[1] = *(const uint4*)(a_s + row * K + kk + hA + 16);
    bf.q[0] = *(const uint4*)(wrow + kk + hB);
    bf.q[1] = *(const uint4*)(wrow + kk + hB + 16);
    acc = __builtin_amdgcn_wmma_f32_16x16x32_bf16(
        false, __builtin_bit_cast(v16bf, af.v),
        false, __builtin_bit_cast(v16bf, bf.v),
        (short)0, acc, false, false);
  }

  const float bval = BIAS ? bias[col] : 0.f;
  const int rbase = m0 + ((lane & 16) ? 8 : 0);
#pragma unroll
  for (int r = 0; r < 8; ++r) {
    const size_t idx = (size_t)(rbase + r) * Ncols + col;
    float v = acc[r] + bval;
    if (RELU) v = fmaxf(v, 0.f);
    if (RES)  v += Res[idx];
    C[idx] = v;
  }
}

// ---------------------------------------------------------------------------
// x = input @ W_embed + b_embed   (K=2, elementwise style)
// ---------------------------------------------------------------------------
__global__ void embed_kernel(const float* __restrict__ inp, const float* __restrict__ We,
                             const float* __restrict__ be, float* __restrict__ X)
{
  int idx = blockIdx.x * 256 + threadIdx.x;
  if (idx >= BN_ * DD) return;
  int m = idx >> 7, d = idx & 127;
  X[idx] = inp[2 * m] * We[d] + inp[2 * m + 1] * We[DD + d] + be[d];
}

// ---------------------------------------------------------------------------
// BatchNorm over (B,N): one block per channel -> mean/var; then apply in place
// ---------------------------------------------------------------------------
__global__ __launch_bounds__(256)
void bn_stats(const float* __restrict__ X, float* __restrict__ stats)
{
  const int c = blockIdx.x;
  float s = 0.f, s2 = 0.f;
  for (int r = threadIdx.x; r < BN_; r += 256) {
    float v = X[(size_t)r * DD + c];
    s += v; s2 += v * v;
  }
  __shared__ float rs[256], rs2[256];
  rs[threadIdx.x] = s; rs2[threadIdx.x] = s2;
  __syncthreads();
  for (int o = 128; o > 0; o >>= 1) {
    if (threadIdx.x < o) { rs[threadIdx.x] += rs[threadIdx.x + o];
                           rs2[threadIdx.x] += rs2[threadIdx.x + o]; }
    __syncthreads();
  }
  if (threadIdx.x == 0) {
    float mean = rs[0] / (float)BN_;
    float var  = rs2[0] / (float)BN_ - mean * mean;   // biased var (jnp.var)
    stats[c] = mean; stats[DD + c] = var;
  }
}

__global__ void bn_apply(float* __restrict__ X, const float* __restrict__ stats,
                         const float* __restrict__ w, const float* __restrict__ b)
{
  int idx = blockIdx.x * 256 + threadIdx.x;
  if (idx >= BN_ * DD) return;
  int c = idx & 127;
  X[idx] = (X[idx] - stats[c]) * rsqrtf(stats[DD + c] + EPS_) * w[c] + b[c];
}

// ---------------------------------------------------------------------------
// Encoder MHA core: per (b,h) block, K/V rows in LDS, f32 online softmax.
// ---------------------------------------------------------------------------
__global__ __launch_bounds__(128)
void attn_kernel(const float* __restrict__ Q, const float* __restrict__ Kt,
                 const float* __restrict__ V, float* __restrict__ O)
{
  const int bh = blockIdx.x;
  const int b = bh >> 3, h = bh & 7;
  __shared__ float k_s[NN * DKK], v_s[NN * DKK];
  const size_t base = (size_t)b * NN * DD + h * DKK;
  for (int i = threadIdx.x; i < NN * DKK; i += 128) {
    int n = i >> 4, d = i & 15;
    k_s[i] = Kt[base + (size_t)n * DD + d];
    v_s[i] = V [base + (size_t)n * DD + d];
  }
  __syncthreads();
  for (int i = threadIdx.x; i < NN; i += 128) {
    float q[DKK];
#pragma unroll
    for (int d = 0; d < DKK; ++d) q[d] = Q[base + (size_t)i * DD + d];
    float m = -3.4e38f, l = 0.f, o[DKK];
#pragma unroll
    for (int d = 0; d < DKK; ++d) o[d] = 0.f;
    for (int n = 0; n < NN; ++n) {
      float s = 0.f;
#pragma unroll
      for (int d = 0; d < DKK; ++d) s += q[d] * k_s[n * DKK + d];
      s *= 0.25f;                                   // 1/sqrt(DK)
      if (s > m) {
        float sc = __expf(m - s);
        l *= sc;
#pragma unroll
        for (int d = 0; d < DKK; ++d) o[d] *= sc;
        m = s;
      }
      float e = __expf(s - m);
      l += e;
#pragma unroll
      for (int d = 0; d < DKK; ++d) o[d] += e * v_s[n * DKK + d];
    }
    float inv = 1.f / l;
#pragma unroll
    for (int d = 0; d < DKK; ++d) O[base + (size_t)i * DD + d] = o[d] * inv;
  }
}

// ---------------------------------------------------------------------------
// Greedy decoder rollout: one block per batch element, 99 sequential steps.
// Kg/Vg/Kl (3 x 100 x 128 f32 = 153.6 KB) live in dynamic LDS (CDNA5 320 KB).
// Running sum gsum replaces repeated masked means; emb[sel] fetched from L2.
// Writes cost[b], log_p[b], tours[b][100] (as floats) into d_out.
// ---------------------------------------------------------------------------
__global__ __launch_bounds__(128)
void decoder_kernel(const float* __restrict__ emb, const float* __restrict__ kvl,
                    const float* __restrict__ coords, const float* __restrict__ Wf,
                    const float* __restrict__ Ws, const float* __restrict__ Wo,
                    float* __restrict__ out)
{
  extern __shared__ float dyn[];
  float* kg_s = dyn;                 // NN*DD
  float* vg_s = kg_s + NN * DD;      // NN*DD
  float* kl_s = vg_s + NN * DD;      // NN*DD
  __shared__ float avail[NN];
  __shared__ float compat[HH * NN];
  __shared__ float ctx[DD], gv[DD], gl[DD], gsum[DD], ef[DD], el[DD];
  __shared__ float logits[NN];
  __shared__ int   tour[NN];
  __shared__ float lp_s;
  __shared__ int   sel_s;

  const int b = blockIdx.x;
  const int tid = threadIdx.x;
  const size_t eb = (size_t)b * NN * DD;

  for (int i = tid; i < NN * DD; i += 128) {
    int n = i >> 7, d = i & 127;
    size_t kb = ((size_t)b * NN + n) * 384;
    kg_s[i] = kvl[kb + d];
    vg_s[i] = kvl[kb + 128 + d];
    kl_s[i] = kvl[kb + 256 + d];
  }
  for (int n = tid; n < NN; n += 128) avail[n] = (n == 0) ? 0.f : 1.f;
  {
    float f0 = emb[eb + tid];
    ef[tid] = f0; el[tid] = f0;
    float s = 0.f;
    for (int n = 1; n < NN; ++n) s += emb[eb + (size_t)n * DD + tid];
    gsum[tid] = s;                                   // node 0 excluded (avail0)
  }
  if (tid == 0) { lp_s = 0.f; tour[0] = 0; }
  __syncthreads();

  for (int step = 0; step < NN - 1; ++step) {
    const float invc = 1.f / (float)(NN - 1 - step);
    // ctx = gmean@W_fixed + [ef,el]@W_step   (thread tid owns output col tid)
    {
      float s = 0.f;
      for (int d = 0; d < DD; ++d) {
        s += gsum[d] * invc * Wf[d * DD + tid];
        s += ef[d] * Ws[d * DD + tid];
        s += el[d] * Ws[(DD + d) * DD + tid];
      }
      ctx[tid] = s;
    }
    __syncthreads();
    // masked per-head compatibilities
    for (int i = tid; i < HH * NN; i += 128) {
      int h = i / NN, n = i - h * NN;
      float s = 0.f;
#pragma unroll
      for (int d = 0; d < DKK; ++d) s += ctx[h * DKK + d] * kg_s[n * DD + h * DKK + d];
      compat[i] = (avail[n] > 0.5f) ? s * 0.25f : NEGV;
    }
    __syncthreads();
    // softmax per head (8 threads, serial over 100)
    if (tid < HH) {
      float mx = NEGV;
      for (int n = 0; n < NN; ++n) mx = fmaxf(mx, compat[tid * NN + n]);
      float s = 0.f;
      for (int n = 0; n < NN; ++n) {
        float e = __expf(compat[tid * NN + n] - mx);
        compat[tid * NN + n] = e; s += e;
      }
      float inv = 1.f / s;
      for (int n = 0; n < NN; ++n) compat[tid * NN + n] *= inv;
    }
    __syncthreads();
    // glimpse vector (h = tid/16, d = tid%16)
    {
      int h = tid >> 4, d = tid & 15;
      float s = 0.f;
      for (int n = 0; n < NN; ++n) s += compat[h * NN + n] * vg_s[n * DD + h * DKK + d];
      gv[tid] = s;
    }
    __syncthreads();
    // glimpse @ W_out
    {
      float s = 0.f;
      for (int d = 0; d < DD; ++d) s += gv[d] * Wo[d * DD + tid];
      gl[tid] = s;
    }
    __syncthreads();
    // clipped-tanh logits
    for (int n = tid; n < NN; n += 128) {
      float s = 0.f;
      for (int d = 0; d < DD; ++d) s += gl[d] * kl_s[n * DD + d];
      s *= 0.0883883476483184f;                      // 1/sqrt(128)
      logits[n] = (avail[n] > 0.5f) ? tanhf(s) * 10.f : NEGV;
    }
    __syncthreads();
    // argmax + log-softmax accumulation (serial, thread 0)
    if (tid == 0) {
      float mx = NEGV; int best = 0;
      for (int n = 0; n < NN; ++n) if (logits[n] > mx) { mx = logits[n]; best = n; }
      float s = 0.f;
      for (int n = 0; n < NN; ++n) s += __expf(logits[n] - mx);
      lp_s += logits[best] - mx - __logf(s);
      sel_s = best; avail[best] = 0.f; tour[step + 1] = best;
    }
    __syncthreads();
    {
      float e = emb[eb + (size_t)sel_s * DD + tid];
      el[tid] = e;
      gsum[tid] -= e;
    }
    __syncthreads();
  }

  if (tid == 0) {
    const size_t cb = (size_t)b * NN * 2;
    float px = coords[cb + tour[0] * 2], py = coords[cb + tour[0] * 2 + 1];
    float fx = px, fy = py, c = 0.f;
    for (int t = 1; t < NN; ++t) {
      float x = coords[cb + tour[t] * 2], y = coords[cb + tour[t] * 2 + 1];
      float dx = x - px, dy = y - py;
      c += sqrtf(dx * dx + dy * dy);
      px = x; py = y;
    }
    float dx = fx - px, dy = fy - py;
    c += sqrtf(dx * dx + dy * dy);
    out[b] = c;
    out[BB + b] = lp_s;
  }
  for (int t = tid; t < NN; t += 128)
    out[2 * BB + (size_t)b * NN + t] = (float)tour[t];
}

// ---------------------------------------------------------------------------
extern "C" void kernel_launch(void* const* d_in, const int* in_sizes, int n_in,
                              void* d_out, int out_size, void* d_ws, size_t ws_size,
                              hipStream_t stream)
{
  (void)in_sizes; (void)n_in; (void)out_size; (void)ws_size;
  const float* inp   = (const float*)d_in[0];
  const float* We    = (const float*)d_in[1];
  const float* be    = (const float*)d_in[2];
  const float* Wq    = (const float*)d_in[3];
  const float* Wk    = (const float*)d_in[4];
  const float* Wv    = (const float*)d_in[5];
  const float* Wo    = (const float*)d_in[6];
  const float* bn1w  = (const float*)d_in[7];
  const float* bn1b  = (const float*)d_in[8];
  const float* bn2w  = (const float*)d_in[9];
  const float* bn2b  = (const float*)d_in[10];
  const float* Wff1  = (const float*)d_in[11];
  const float* bff1  = (const float*)d_in[12];
  const float* Wff2  = (const float*)d_in[13];
  const float* bff2  = (const float*)d_in[14];
  const float* Wnode = (const float*)d_in[15];
  const float* Wfix  = (const float*)d_in[16];
  const float* Wstep = (const float*)d_in[17];
  const float* Wout  = (const float*)d_in[18];
  // d_in[19] = n_rollout, assumed 1 (setup value); grid dims can't depend on
  // a device scalar under graph capture.

  float* ws    = (float*)d_ws;
  float* xb    = ws;                                  // BN*128
  float* bufA  = xb   + (size_t)BN_ * DD;             // BN*512 (q|k|v|ao, ff1)
  float* bufB  = bufA + (size_t)BN_ * HFF_;           // BN*128 (h)
  float* kvl   = bufB + (size_t)BN_ * DD;             // BN*384
  float* stats = kvl  + (size_t)BN_ * 384;            // 256
  float* q  = bufA;
  float* k  = bufA + 1 * (size_t)BN_ * DD;
  float* v  = bufA + 2 * (size_t)BN_ * DD;
  float* ao = bufA + 3 * (size_t)BN_ * DD;

  // bf16 transposed weight cache (one-time converts, graph-safe/deterministic)
  unsigned short* wqT   = (unsigned short*)(stats + 256);   // 3 * 128*128
  unsigned short* wkT   = wqT   + 3 * DD * DD;
  unsigned short* wvT   = wkT   + 3 * DD * DD;
  unsigned short* woT   = wvT   + 3 * DD * DD;
  unsigned short* ff1T  = woT   + 3 * DD * DD;              // 3 * 128*512
  unsigned short* ff2T  = ff1T  + 3 * DD * HFF_;            // 3 * 512*128
  unsigned short* nodeT = ff2T  + 3 * HFF_ * DD;            // 384*128

  const int gdd  = (DD * DD + 255) / 256;
  const int gdf  = (DD * HFF_ + 255) / 256;
  for (int l = 0; l < LL; ++l) {
    convT_kernel<<<gdd, 256, 0, stream>>>(Wq  + (size_t)l*DD*DD,   wqT  + (size_t)l*DD*DD,   DD,  DD);
    convT_kernel<<<gdd, 256, 0, stream>>>(Wk  + (size_t)l*DD*DD,   wkT  + (size_t)l*DD*DD,   DD,  DD);
    convT_kernel<<<gdd, 256, 0, stream>>>(Wv  + (size_t)l*DD*DD,   wvT  + (size_t)l*DD*DD,   DD,  DD);
    convT_kernel<<<gdd, 256, 0, stream>>>(Wo  + (size_t)l*DD*DD,   woT  + (size_t)l*DD*DD,   DD,  DD);
    convT_kernel<<<gdf, 256, 0, stream>>>(Wff1 + (size_t)l*DD*HFF_, ff1T + (size_t)l*DD*HFF_, DD,  HFF_);
    convT_kernel<<<gdf, 256, 0, stream>>>(Wff2 + (size_t)l*HFF_*DD, ff2T + (size_t)l*HFF_*DD, HFF_, DD);
  }
  convT_kernel<<<(DD*384 + 255)/256, 256, 0, stream>>>(Wnode, nodeT, DD, 384);

  const int elems = BN_ * DD;
  const int eg = (elems + 255) / 256;
  const int MT = BN_ / 16;                              // 6400 row tiles

  embed_kernel<<<eg, 256, 0, stream>>>(inp, We, be, xb);
  for (int l = 0; l < LL; ++l) {
    gemm_wmma<DD, false, false, false><<<dim3(MT, 1), 256, 0, stream>>>(
        xb, wqT + (size_t)l*DD*DD, nullptr, nullptr, q, DD);
    gemm_wmma<DD, false, false, false><<<dim3(MT, 1), 256, 0, stream>>>(
        xb, wkT + (size_t)l*DD*DD, nullptr, nullptr, k, DD);
    gemm_wmma<DD, false, false, false><<<dim3(MT, 1), 256, 0, stream>>>(
        xb, wvT + (size_t)l*DD*DD, nullptr, nullptr, v, DD);
    attn_kernel<<<BB * HH, 128, 0, stream>>>(q, k, v, ao);
    gemm_wmma<DD, false, false, true><<<dim3(MT, 1), 256, 0, stream>>>(
        ao, woT + (size_t)l*DD*DD, nullptr, xb, bufB, DD);
    bn_stats<<<DD, 256, 0, stream>>>(bufB, stats);
    bn_apply<<<eg, 256, 0, stream>>>(bufB, stats, bn1w + l * DD, bn1b + l * DD);
    gemm_wmma<DD, true, true, false><<<dim3(MT, 4), 256, 0, stream>>>(
        bufB, ff1T + (size_t)l*DD*HFF_, bff1 + l * HFF_, nullptr, bufA, HFF_);
    gemm_wmma<HFF_, true, false, true><<<dim3(MT, 1), 256, 0, stream>>>(
        bufA, ff2T + (size_t)l*HFF_*DD, bff2 + l * DD, bufB, xb, DD);
    bn_stats<<<DD, 256, 0, stream>>>(xb, stats);
    bn_apply<<<eg, 256, 0, stream>>>(xb, stats, bn2w + l * DD, bn2b + l * DD);
  }
  gemm_wmma<DD, false, false, false><<<dim3(MT, 3), 256, 0, stream>>>(
      xb, nodeT, nullptr, nullptr, kvl, 384);
  decoder_kernel<<<BB, 128, 3 * NN * DD * (int)sizeof(float), stream>>>(
      xb, kvl, inp, Wfix, Wstep, Wout, (float*)d_out);
}
